// GCN_miRNA_85341000171598
// MI455X (gfx1250) — compile-verified, matrix operand order checked by reading, not run on hardware
//
#include <hip/hip_runtime.h>

// ---------------------------------------------------------------------------
// GCN (2x GCNConv + mean pool) for MI455X / gfx1250, wave32, WMMA bf16->f32.
// GEMM stages tiles into LDS in *fragment order*; staging packs 8 bf16 per
// ds_store_b128, compute loads whole v16bf fragments with 2x ds_load_b128.
// ---------------------------------------------------------------------------

typedef __attribute__((ext_vector_type(16))) __bf16 v16bf;
typedef __attribute__((ext_vector_type(8)))  __bf16 v8bf;
typedef __attribute__((ext_vector_type(8)))  float  v8f;

#define N_NODES  50000
#define N_GRAPHS 128
#define IN_DIM   640
#define HID      128
#define N_EDGES  800000

// ---------------------------------------------------------------------------
// Init: deg = 1.0 (self loop), zero edge-accumulator, zero pool buffers.
// ---------------------------------------------------------------------------
__global__ void init_kernel(float* __restrict__ deg, float* __restrict__ acc,
                            float* __restrict__ sums, float* __restrict__ cnts) {
    const long long stride = (long long)gridDim.x * blockDim.x;
    const long long i0 = (long long)blockIdx.x * blockDim.x + threadIdx.x;
    if (i0 < N_NODES) deg[i0] = 1.0f;                 // self-loop contribution
    if (i0 < N_GRAPHS * HID) sums[i0] = 0.0f;
    if (i0 < N_GRAPHS) cnts[i0] = 0.0f;
    for (long long j = i0; j < (long long)N_NODES * HID; j += stride)
        acc[j] = 0.0f;
}

__global__ void zero_kernel(float* __restrict__ p, long long n) {
    const long long stride = (long long)gridDim.x * blockDim.x;
    for (long long j = (long long)blockIdx.x * blockDim.x + threadIdx.x; j < n; j += stride)
        p[j] = 0.0f;
}

// ---------------------------------------------------------------------------
// In-degree (incl. self loop already init'd to 1): deg[col[e]] += 1
// ---------------------------------------------------------------------------
__global__ void deg_kernel(const long long* __restrict__ ei, float* __restrict__ deg) {
    const int e = blockIdx.x * blockDim.x + threadIdx.x;
    if (e < N_EDGES) {
        const int c = (int)ei[(long long)N_EDGES + e];
        atomicAdd(&deg[c], 1.0f);
    }
}

__global__ void dinv_kernel(float* __restrict__ deg) {
    const int i = blockIdx.x * blockDim.x + threadIdx.x;
    if (i < N_NODES) deg[i] = rsqrtf(fmaxf(deg[i], 1.0f));
}

// ---------------------------------------------------------------------------
// Tiled GEMM  Y[M x 128] = X[M x K] * W[K x 128]  (fp32 in, bf16 WMMA, f32 acc)
// Block = 256 threads = 8 waves -> one 16-row x 128-col strip; wave w owns the
// 16x16 tile at cols [16w,16w+16). K stepped by 64 (two WMMAs per barrier).
//
// ISA 7.12.2 16-bit operand swizzle: lanes 0-15 hold K={0..7,16..23}, lanes
// 16-31 hold K={8..15,24..31}. A k-group of 8 consecutive K (g*8..g*8+7) maps
// to ONE lane-half and 8 CONTIGUOUS fragment elements:
//   chunk ch = g>>2, lane-half lh = g&1, element base = 8*((g>>1)&1)
// -> stage each k-group as a single packed ds_store_b128.
// Requires K % 64 == 0 (640, 128 OK) and M % 16 == 0 (50000 = 3125*16).
// ---------------------------------------------------------------------------
__global__ __launch_bounds__(256)
void gemm_bf16_wmma(const float* __restrict__ X, const float* __restrict__ W,
                    float* __restrict__ Y, int K) {
    // Fragment-ordered staging: [chunk][lane][16 elems], 32 B per lane fragment
    __shared__ alignas(32) __bf16 Afrag[2 * 32 * 16];        //  2 KB
    __shared__ alignas(32) __bf16 Bfrag[2 * 8 * 32 * 16];    // 16 KB

    const int tid  = threadIdx.x;
    const int wave = tid >> 5;
    const int lane = tid & 31;
    const int rowBase = blockIdx.x * 16;
    const int half = lane >> 4;
    const int m    = lane & 15;

    v8f c = {};
    for (int k0 = 0; k0 < K; k0 += 64) {
        // ---- Stage A (16 rows x 64 K): 128 k-groups, threads 0-127 ----
        if (tid < 128) {
            const int r = tid >> 3, g = tid & 7;
            const int ch = g >> 2, lh = g & 1, ebase = 8 * ((g >> 1) & 1);
            const float* src = X + (long long)(rowBase + r) * K + k0 + g * 8;
            const float4 x0 = *(const float4*)src;          // global_load_b128
            const float4 x1 = *(const float4*)(src + 4);    // global_load_b128
            v8bf t;
            t[0] = (__bf16)x0.x; t[1] = (__bf16)x0.y;
            t[2] = (__bf16)x0.z; t[3] = (__bf16)x0.w;
            t[4] = (__bf16)x1.x; t[5] = (__bf16)x1.y;
            t[6] = (__bf16)x1.z; t[7] = (__bf16)x1.w;
            *(v8bf*)(Afrag + ch * 512 + (lh * 16 + r) * 16 + ebase) = t;
        }
        // ---- Stage B (64 K x 128 n): 1024 k-groups, 4 per thread ----
#pragma unroll
        for (int t4 = 0; t4 < 4; ++t4) {
            const int idx = tid + t4 * 256;       // [0,1024)
            const int g = idx >> 7, n = idx & 127;
            const int ch = g >> 2, lh = g & 1, ebase = 8 * ((g >> 1) & 1);
            const int w = n >> 4, lm = n & 15;
            const float* src = W + (long long)(k0 + g * 8) * HID + n;
            v8bf t;
#pragma unroll
            for (int j = 0; j < 8; ++j) t[j] = (__bf16)src[(long long)j * HID];
            *(v8bf*)(Bfrag + ch * 4096 + (w * 32 + lh * 16 + lm) * 16 + ebase) = t;
        }
        __syncthreads();

        // ---- Two WMMAs per staged super-tile: fragments load as 2x b128 ----
#pragma unroll
        for (int ch = 0; ch < 2; ++ch) {
            const v16bf a = *(const v16bf*)(Afrag + ch * 512  + lane * 16);
            const v16bf b = *(const v16bf*)(Bfrag + ch * 4096 + (wave * 32 + lane) * 16);
            c = __builtin_amdgcn_wmma_f32_16x16x32_bf16(
                    /*neg_a=*/false, a, /*neg_b=*/false, b,
                    /*c_mod=*/(short)0, c, /*reuse_a=*/false, /*reuse_b=*/false);
        }
        __syncthreads();
    }

    // C/D layout: VGPR r -> M=r (lanes 0-15), M=r+8 (lanes 16-31); N = lane%16
#pragma unroll
    for (int r = 0; r < 8; ++r) {
        const int row = rowBase + r + 8 * half;
        Y[(long long)row * HID + wave * 16 + m] = c[r];
    }
}

// ---------------------------------------------------------------------------
// Edge-parallel message passing: one wave per edge, 4 features per lane.
// acc[col] += xw[row] * (dinv[row]*dinv[col])
// ---------------------------------------------------------------------------
__global__ __launch_bounds__(256)
void msg_kernel(const long long* __restrict__ ei, const float* __restrict__ xw,
                const float* __restrict__ dinv, float* __restrict__ acc) {
    const long long t = (long long)blockIdx.x * blockDim.x + threadIdx.x;
    const long long e = t >> 5;
    const int lane = (int)(t & 31);
    if (e >= N_EDGES) return;
    const int r = (int)ei[e];
    const int c = (int)ei[(long long)N_EDGES + e];
    const float norm = dinv[r] * dinv[c];

    const float* src = xw + (long long)r * HID + lane * 4;
    __builtin_prefetch(src, 0, 0);                     // global_prefetch_b8
    const float4 v = *(const float4*)src;
    float* dst = acc + (long long)c * HID + lane * 4;
    atomicAdd(dst + 0, v.x * norm);
    atomicAdd(dst + 1, v.y * norm);
    atomicAdd(dst + 2, v.z * norm);
    atomicAdd(dst + 3, v.w * norm);
}

// ---------------------------------------------------------------------------
// Post: out = (acc + xw * dinv^2 + bias), optional ReLU.  (self-loop term)
// ---------------------------------------------------------------------------
__global__ void post_kernel(const float* __restrict__ acc, const float* __restrict__ xw,
                            const float* __restrict__ dinv, const float* __restrict__ bias,
                            float* __restrict__ out, int do_relu) {
    const long long i = (long long)blockIdx.x * blockDim.x + threadIdx.x;
    if (i >= (long long)N_NODES * HID) return;
    const int node = (int)(i >> 7);
    const int f = (int)(i & 127);
    const float d = dinv[node];
    float v = acc[i] + xw[i] * d * d + bias[f];
    out[i] = do_relu ? fmaxf(v, 0.0f) : v;
}

// ---------------------------------------------------------------------------
// Mean pool (scatter + finalize)
// ---------------------------------------------------------------------------
__global__ void pool_scatter(const float* __restrict__ h, const long long* __restrict__ batch,
                             float* __restrict__ sums, float* __restrict__ cnts) {
    const long long i = (long long)blockIdx.x * blockDim.x + threadIdx.x;
    if (i >= (long long)N_NODES * HID) return;
    const int node = (int)(i >> 7);
    const int f = (int)(i & 127);
    const int g = (int)batch[node];
    atomicAdd(&sums[g * HID + f], h[i]);
    if (f == 0) atomicAdd(&cnts[g], 1.0f);
}

__global__ void pool_div(const float* __restrict__ sums, const float* __restrict__ cnts,
                         float* __restrict__ out) {
    const int i = blockIdx.x * blockDim.x + threadIdx.x;
    if (i < N_GRAPHS * HID)
        out[i] = sums[i] / fmaxf(cnts[i >> 7], 1.0f);
}

// ---------------------------------------------------------------------------
// Launcher
// ---------------------------------------------------------------------------
extern "C" void kernel_launch(void* const* d_in, const int* in_sizes, int n_in,
                              void* d_out, int out_size, void* d_ws, size_t ws_size,
                              hipStream_t stream) {
    const float*     x_emb = (const float*)d_in[0];
    const long long* ei    = (const long long*)d_in[1];   // int64 edge_index [2, E]
    const long long* batch = (const long long*)d_in[2];   // int64 [N]
    const float*     W1    = (const float*)d_in[3];
    const float*     b1    = (const float*)d_in[4];
    const float*     W3    = (const float*)d_in[5];
    const float*     b3    = (const float*)d_in[6];
    float* out = (float*)d_out;

    // Workspace layout (floats)
    const long long NH = (long long)N_NODES * HID;   // 6.4M
    float* ws      = (float*)d_ws;
    float* buf_xw  = ws;                    // [N, HID]  X@W   (reused both layers)
    float* buf_acc = ws + NH;               // [N, HID]  edge scatter accumulator
    float* buf_h1  = ws + 2 * NH;           // [N, HID]  layer-1 activations
    float* dinv    = ws + 3 * NH;           // [N]       deg -> dinv (in place)
    float* sums    = dinv + N_NODES;        // [G, HID]
    float* cnts    = sums + N_GRAPHS * HID; // [G]

    const int T = 256;
    const int gNode = (N_NODES + T - 1) / T;
    const int gEdge = (N_EDGES + T - 1) / T;
    const int gNH   = (int)((NH + T - 1) / T);
    const int gMsg  = (int)(((long long)N_EDGES * 32 + T - 1) / T);
    const int gGemm = N_NODES / 16;         // 3125

    // Degree / normalization
    init_kernel<<<gNode, T, 0, stream>>>(dinv, buf_acc, sums, cnts);
    deg_kernel<<<gEdge, T, 0, stream>>>(ei, dinv);
    dinv_kernel<<<gNode, T, 0, stream>>>(dinv);

    // ---- Layer 1 ----
    gemm_bf16_wmma<<<gGemm, T, 0, stream>>>(x_emb, W1, buf_xw, IN_DIM);
    msg_kernel<<<gMsg, T, 0, stream>>>(ei, buf_xw, dinv, buf_acc);
    post_kernel<<<gNH, T, 0, stream>>>(buf_acc, buf_xw, dinv, b1, buf_h1, /*relu=*/1);

    // ---- Layer 2 ----
    zero_kernel<<<gNode, T, 0, stream>>>(buf_acc, NH);
    gemm_bf16_wmma<<<gGemm, T, 0, stream>>>(buf_h1, W3, buf_xw, HID);
    msg_kernel<<<gMsg, T, 0, stream>>>(ei, buf_xw, dinv, buf_acc);
    post_kernel<<<gNH, T, 0, stream>>>(buf_acc, buf_xw, dinv, b3, buf_acc, /*relu=*/0);

    // ---- Mean pool ----
    pool_scatter<<<gNH, T, 0, stream>>>(buf_acc, batch, sums, cnts);
    pool_div<<<(N_GRAPHS * HID + T - 1) / T, T, 0, stream>>>(sums, cnts, out);
}
